// CausalSelfAttention_21449066676664
// MI455X (gfx1250) — matrix operand order repaired and checked by reading, not compiled, not run on hardware
//
#include <hip/hip_runtime.h>
#include <stdint.h>

typedef __attribute__((ext_vector_type(16))) __bf16 v16bf;
typedef __attribute__((ext_vector_type(8)))  __bf16 v8bf;
typedef __attribute__((ext_vector_type(8)))  float  v8f;
typedef __attribute__((ext_vector_type(4)))  unsigned int v4u;
typedef __attribute__((ext_vector_type(8)))  int v8i;
typedef __attribute__((ext_vector_type(4)))  int v4i;

constexpr int kB    = 4;
constexpr int kT    = 2048;
constexpr int kDIM  = 1024;
constexpr int kH    = 16;
constexpr int kDh   = 64;
constexpr int kINNER = kH * kDh;            // 1024
constexpr float kSCALE = 0.125f;            // 64^-0.5

// ---- helpers -------------------------------------------------------------

// Load a 16-element bf16 operand register from two 16B chunks (global or LDS).
static __device__ __forceinline__ v16bf ld16(const __bf16* p_lo, const __bf16* p_hi) {
  v8bf lo = *reinterpret_cast<const v8bf*>(p_lo);
  v8bf hi = *reinterpret_cast<const v8bf*>(p_hi);
  return __builtin_shufflevector(lo, hi, 0,1,2,3,4,5,6,7,8,9,10,11,12,13,14,15);
}

static __device__ __forceinline__ v8f wmma_bf16(v16bf a, v16bf b, v8f c) {
  // D = A(16x32 bf16) * B(32x16 bf16) + C(16x16 f32)
  return __builtin_amdgcn_wmma_f32_16x16x32_bf16(false, a, false, b, (short)0, c, false, false);
}

// LDS byte offset of a __shared__-derived pointer: low 32 bits of the
// flat shared-aperture address are the LDS offset (ISA aperture mapping).
static __device__ __forceinline__ unsigned lds_off(const void* p) {
  return (unsigned)(uintptr_t)p;
}

// TDM 2D tile load: tile = [64 rows][32 elems] of bf16, row stride
// rowStrideElems, from a tensor of tensorRows x tensorK elements.
// D# layout per CDNA5 ISA ch.8: group0 = flags/lds_addr/global_addr,
// group1 = dims/tile/strides; groups 2-4 zero => 2D tensor.
static __device__ __forceinline__ void tdm_load_2d(const __bf16* gsrc, unsigned ldsByteOff,
                                                   unsigned tensorK, unsigned tensorRows,
                                                   unsigned rowStrideElems) {
  unsigned long long ga = (unsigned long long)(uintptr_t)gsrc;
  v4u g0 = {};
  g0[0] = 1u;                                            // count=1 (valid), user mode
  g0[1] = ldsByteOff;                                    // lds_addr
  g0[2] = (unsigned)ga;                                  // global_addr[31:0]
  g0[3] = (unsigned)((ga >> 32) & 0x01ffffffu) | (2u << 30); // addr[56:32] | type=2
  v8i g1 = {};
  g1[0] = (int)(1u << 16);                               // data_size=1 (2 bytes)
  g1[1] = (int)((tensorK & 0xffffu) << 16);              // tensor_dim0[15:0] @ bits 63:48
  g1[2] = (int)(((tensorK >> 16) & 0xffffu) | ((tensorRows & 0xffffu) << 16));
  g1[3] = (int)(((tensorRows >> 16) & 0xffffu) | (32u << 16)); // tile_dim0 = 32
  g1[4] = 64;                                            // tile_dim1 = 64 (tile_dim2 = 0)
  g1[5] = (int)rowStrideElems;                           // tensor_dim0_stride[31:0]
  v4i z4 = {};
  v8i z8 = {};
  __builtin_amdgcn_tensor_load_to_lds(g0, g1, z4, z4, z8, 0);
}

// ---- kernel 0: f32 -> bf16 conversion ------------------------------------

__global__ void cvt_f32_bf16(const float* __restrict__ src, __bf16* __restrict__ dst, int n) {
  int i = blockIdx.x * blockDim.x + threadIdx.x;
  int stride = gridDim.x * blockDim.x;
  for (; i < n; i += stride) dst[i] = (__bf16)src[i];
}

// ---- kernel 1: QKV projection GEMM with TDM-staged weights ---------------
// Block = 8 waves: 256 rows x 64 cols. Per k-step the 64x32 weight tile is
// DMA'd into LDS by the TDM (double-buffered), shared by all 8 waves.

__global__ void __launch_bounds__(256)
__attribute__((amdgpu_waves_per_eu(2, 4)))
qkv_gemm(const __bf16* __restrict__ A, const __bf16* __restrict__ W,
         __bf16* __restrict__ qws, __bf16* __restrict__ kws,
         __bf16* __restrict__ vtws) {
  __shared__ __align__(16) __bf16 bsh[2][64 * 32];

  const int lane = threadIdx.x & 31;
  const int wv   = threadIdx.x >> 5;
  const int colBlocks = (3 * kINNER) / 64;      // 48
  const int rb = (blockIdx.x / colBlocks) * 256 + wv * 32;
  const int cb = (blockIdx.x % colBlocks) * 64;
  const int n  = lane & 15, hi = lane >> 4;

  v8f acc[2][4] = {};

  const __bf16* arow0 = A + (size_t)(rb + n) * kDIM;
  const __bf16* arow1 = A + (size_t)(rb + 16 + n) * kDIM;
  const __bf16* wtile = W + (size_t)cb * kDIM;  // 64 rows (cols of C) x K

  constexpr int nSteps = kDIM / 32;             // 32
  if (wv == 0)
    tdm_load_2d(wtile, lds_off(&bsh[0][0]), kDIM, 3 * kINNER, kDIM);

  #pragma unroll 1
  for (int i = 0; i < nSteps; ++i) {
    const int kk = i * 32;
    if (wv == 0) {
      if (i + 1 < nSteps) {
        tdm_load_2d(wtile + kk + 32, lds_off(&bsh[(i + 1) & 1][0]), kDIM, 3 * kINNER, kDIM);
        __builtin_amdgcn_s_wait_tensorcnt(1);   // buffer i complete
      } else {
        __builtin_amdgcn_s_wait_tensorcnt(0);
      }
    }
    __syncthreads();                            // publish buffer i

    __builtin_prefetch(arow0 + kk + 64, 0, 1);
    v16bf a0 = ld16(arow0 + kk + 8 * hi, arow0 + kk + 16 + 8 * hi);
    v16bf a1 = ld16(arow1 + kk + 8 * hi, arow1 + kk + 16 + 8 * hi);
    const __bf16* bb = &bsh[i & 1][0];
    #pragma unroll
    for (int j = 0; j < 4; ++j) {
      const __bf16* bp = bb + (j * 16 + n) * 32 + 8 * hi;
      v16bf b = ld16(bp, bp + 16);
      acc[0][j] = wmma_bf16(a0, b, acc[0][j]);
      acc[1][j] = wmma_bf16(a1, b, acc[1][j]);
    }
    __syncthreads();                            // done with buffer i before overwrite
  }

  // C layout: lane -> col n (L%16); VGPR r -> row r + 8*(L/16).
  #pragma unroll
  for (int i = 0; i < 2; ++i)
    #pragma unroll
    for (int j = 0; j < 4; ++j)
      #pragma unroll
      for (int r = 0; r < 8; ++r) {
        int row = rb + i * 16 + hi * 8 + r;       // global token index
        int col = cb + j * 16 + n;                // 0..3071
        __bf16 v = (__bf16)acc[i][j][r];
        int bb2 = row / kT, t = row % kT;
        int sec = col >> 10, rem = col & 1023;
        int h = rem >> 6, d = rem & 63;
        size_t bh = (size_t)bb2 * kH + h;
        if (sec == 0)      qws[(bh * kT + t) * kDh + d] = v;
        else if (sec == 1) kws[(bh * kT + t) * kDh + d] = v;
        else               vtws[(bh * kDh + d) * kT + t] = v;   // V stored transposed
      }
}

// ---- kernel 2: flash attention (transposed-score formulation) ------------
// One wave per (b, h, 16-row q tile). S^T = K*Q^T, online softmax, O^T = V^T*P^T.

__global__ void attn_kernel(const __bf16* __restrict__ qws, const __bf16* __restrict__ kws,
                            const __bf16* __restrict__ vtws, const uint8_t* __restrict__ maskp,
                            __bf16* __restrict__ ows) {
  const int lane = threadIdx.x & 31;
  const int wv   = threadIdx.x >> 5;
  const int gw   = blockIdx.x * (blockDim.x >> 5) + wv;
  const int qTiles = kT / 16;                   // 128
  const int qt = gw % qTiles;
  const int bh = gw / qTiles;
  const int bb = bh / kH, h = bh % kH;
  const int n  = lane & 15, hi = lane >> 4;
  const int q0 = qt * 16;
  const int qrow = q0 + n;                      // this lane's q row (softmax row)

  // Q as B-operand (table order): elements 0..7 = d 8*hi.., 8..15 = d 16+8*hi..
  const __bf16* qp = qws + ((size_t)bh * kT + qrow) * kDh;
  v16bf bq0 = ld16(qp + 8 * hi,      qp + 16 + 8 * hi);        // d 0..31
  v16bf bq1 = ld16(qp + 32 + 8 * hi, qp + 48 + 8 * hi);   // d 32..63

  v8f oacc[4] = {};                             // O^T: 4 d-tiles of 16, N = q
  float m = -1e30f, l = 0.f;
  const uint8_t* mrow = maskp + (size_t)bb * kT;

  for (int kc = 0; kc < q0 + 16; kc += 32) {
    // S^T tiles: A = K rows (16 keys x 64 d, split into two K=32 WMMAs)
    const __bf16* kp0 = kws + ((size_t)bh * kT + (kc + n)) * kDh;
    const __bf16* kp1 = kws + ((size_t)bh * kT + (kc + 16 + n)) * kDh;
    v8f c0 = {}, c1 = {};
    c0 = wmma_bf16(ld16(kp0 + 8 * hi,      kp0 + 16 + 8 * hi), bq0, c0);
    c0 = wmma_bf16(ld16(kp0 + 32 + 8 * hi, kp0 + 48 + 8 * hi), bq1, c0);
    c1 = wmma_bf16(ld16(kp1 + 8 * hi,      kp1 + 16 + 8 * hi), bq0, c1);
    c1 = wmma_bf16(ld16(kp1 + 32 + 8 * hi, kp1 + 48 + 8 * hi), bq1, c1);

    // scale + causal/padding mask; per-lane rows of softmax = fixed q = qrow
    float sv0[8], sv1[8];
    float mloc = -1e30f;
    #pragma unroll
    for (int r = 0; r < 8; ++r) {
      int k0 = kc + hi * 8 + r;                 // key for c0[r]
      int k1 = k0 + 16;                         // key for c1[r]
      float s0 = c0[r] * kSCALE;
      float s1 = c1[r] * kSCALE;
      if (k0 > qrow || !mrow[k0]) s0 = -1e30f;
      if (k1 > qrow || !mrow[k1]) s1 = -1e30f;
      sv0[r] = s0; sv1[r] = s1;
      mloc = fmaxf(mloc, fmaxf(s0, s1));
    }
    // each q row lives in lanes {n, n+16}: combine across the half-waves
    mloc = fmaxf(mloc, __shfl_xor(mloc, 16, 32));
    float mnew  = fmaxf(m, mloc);
    float scale = __expf(m - mnew);
    float p0[8], p1[8], ls = 0.f;
    #pragma unroll
    for (int r = 0; r < 8; ++r) {
      p0[r] = __expf(sv0[r] - mnew);
      p1[r] = __expf(sv1[r] - mnew);
      ls += p0[r] + p1[r];
    }
    l = l * scale + ls + __shfl_xor(ls, 16, 32);
    m = mnew;
    #pragma unroll
    for (int t = 0; t < 4; ++t)
      #pragma unroll
      for (int r = 0; r < 8; ++r) oacc[t][r] *= scale;

    // Build P^T as B-operand: lane hi=0 needs chunk keys 0..15, hi=1 keys 16..31.
    // Own lane holds half of each tile; partner (xor 16) holds the other half.
    v16bf pB;
    #pragma unroll
    for (int r = 0; r < 8; ++r) {
      float o0 = __shfl_xor(p0[r], 16, 32);
      float o1 = __shfl_xor(p1[r], 16, 32);
      pB[r]     = (__bf16)(hi ? o1 : p0[r]);    // keys 0..7  | 16..23
      pB[r + 8] = (__bf16)(hi ? p1[r] : o0);    // keys 8..15 | 24..31
    }

    // O^T += V^T * P^T : A = V^T rows (d), K = 32 keys, contiguous along t
    #pragma unroll
    for (int t = 0; t < 4; ++t) {
      const __bf16* vp = vtws + ((size_t)bh * kDh + (t * 16 + n)) * kT + kc;
      v16bf va = ld16(vp + 16 * hi, vp + 16 * hi + 8);
      oacc[t] = wmma_bf16(va, pB, oacc[t]);
    }
  }

  // O^T C-layout: lane -> q col; VGPR r -> d = t*16 + 8*hi + r (8 consecutive)
  float inv = 1.0f / l;
  #pragma unroll
  for (int t = 0; t < 4; ++t) {
    v8bf ov;
    #pragma unroll
    for (int r = 0; r < 8; ++r) ov[r] = (__bf16)(oacc[t][r] * inv);
    __bf16* dst = ows + ((size_t)(bb * kT + qrow)) * kINNER + h * kDh + t * 16 + hi * 8;
    *reinterpret_cast<v8bf*>(dst) = ov;
  }
}

// ---- kernel 3: output projection GEMM with TDM-staged weights ------------

__global__ void __launch_bounds__(256)
__attribute__((amdgpu_waves_per_eu(2, 4)))
out_gemm(const __bf16* __restrict__ A, const __bf16* __restrict__ W,
         float* __restrict__ out) {
  __shared__ __align__(16) __bf16 bsh[2][64 * 32];

  const int lane = threadIdx.x & 31;
  const int wv   = threadIdx.x >> 5;
  const int colBlocks = kDIM / 64;              // 16
  const int rb = (blockIdx.x / colBlocks) * 256 + wv * 32;
  const int cb = (blockIdx.x % colBlocks) * 64;
  const int n  = lane & 15, hi = lane >> 4;

  v8f acc[2][4] = {};
  const __bf16* arow0 = A + (size_t)(rb + n) * kINNER;
  const __bf16* arow1 = A + (size_t)(rb + 16 + n) * kINNER;
  const __bf16* wtile = W + (size_t)cb * kINNER;

  constexpr int nSteps = kINNER / 32;           // 32
  if (wv == 0)
    tdm_load_2d(wtile, lds_off(&bsh[0][0]), kINNER, kDIM, kINNER);

  #pragma unroll 1
  for (int i = 0; i < nSteps; ++i) {
    const int kk = i * 32;
    if (wv == 0) {
      if (i + 1 < nSteps) {
        tdm_load_2d(wtile + kk + 32, lds_off(&bsh[(i + 1) & 1][0]), kINNER, kDIM, kINNER);
        __builtin_amdgcn_s_wait_tensorcnt(1);
      } else {
        __builtin_amdgcn_s_wait_tensorcnt(0);
      }
    }
    __syncthreads();

    __builtin_prefetch(arow0 + kk + 64, 0, 1);
    v16bf a0 = ld16(arow0 + kk + 8 * hi, arow0 + kk + 16 + 8 * hi);
    v16bf a1 = ld16(arow1 + kk + 8 * hi, arow1 + kk + 16 + 8 * hi);
    const __bf16* bb = &bsh[i & 1][0];
    #pragma unroll
    for (int j = 0; j < 4; ++j) {
      const __bf16* bp = bb + (j * 16 + n) * 32 + 8 * hi;
      v16bf b = ld16(bp, bp + 16);
      acc[0][j] = wmma_bf16(a0, b, acc[0][j]);
      acc[1][j] = wmma_bf16(a1, b, acc[1][j]);
    }
    __syncthreads();
  }

  #pragma unroll
  for (int i = 0; i < 2; ++i)
    #pragma unroll
    for (int j = 0; j < 4; ++j)
      #pragma unroll
      for (int r = 0; r < 8; ++r) {
        int row = rb + i * 16 + hi * 8 + r;
        int col = cb + j * 16 + n;
        out[(size_t)row * kDIM + col] = acc[i][j][r];
      }
}

// ---- host-side launcher ---------------------------------------------------

extern "C" void kernel_launch(void* const* d_in, const int* in_sizes, int n_in,
                              void* d_out, int out_size, void* d_ws, size_t ws_size,
                              hipStream_t stream) {
  (void)in_sizes; (void)n_in; (void)out_size; (void)ws_size;
  const float*   x     = (const float*)d_in[0];
  const uint8_t* mask  = (const uint8_t*)d_in[1];
  const float*   w_qkv = (const float*)d_in[2];
  const float*   w_out = (const float*)d_in[3];
  float*         out   = (float*)d_out;

  // workspace layout (bytes)
  char* ws = (char*)d_ws;
  const size_t nTok = (size_t)kB * kT;                       // 8192
  __bf16* x_bf    = (__bf16*)(ws);                           // 16.78 MB
  __bf16* wqkv_bf = (__bf16*)(ws + 16777216);                //  6.29 MB
  __bf16* wout_bf = (__bf16*)(ws + 23068672);                //  2.10 MB
  __bf16* q_ws    = (__bf16*)(ws + 25165824);                // [B,H,T,64]
  __bf16* k_ws    = (__bf16*)(ws + 41943040);                // [B,H,T,64]
  __bf16* vT_ws   = (__bf16*)(ws + 58720256);                // [B,H,64,T]
  __bf16* attn_ws = (__bf16*)(ws + 75497472);                // [B,T,1024]

  hipLaunchKernelGGL(cvt_f32_bf16, dim3(2048), dim3(256), 0, stream,
                     x, x_bf, (int)(nTok * kDIM));
  hipLaunchKernelGGL(cvt_f32_bf16, dim3(1024), dim3(256), 0, stream,
                     w_qkv, wqkv_bf, 3 * kINNER * kDIM);
  hipLaunchKernelGGL(cvt_f32_bf16, dim3(512), dim3(256), 0, stream,
                     w_out, wout_bf, kDIM * kINNER);

  // QKV GEMM: 32 row-blocks x 48 col-blocks, 256 threads (8 waves)
  hipLaunchKernelGGL(qkv_gemm, dim3(1536), dim3(256), 0, stream,
                     x_bf, wqkv_bf, q_ws, k_ws, vT_ws);

  // attention: B*H*(T/16) = 8192 waves, 8 waves/block
  hipLaunchKernelGGL(attn_kernel, dim3(1024), dim3(256), 0, stream,
                     q_ws, k_ws, vT_ws, mask, attn_ws);

  // output projection: 32 row-blocks x 16 col-blocks
  hipLaunchKernelGGL(out_gemm, dim3(512), dim3(256), 0, stream,
                     attn_ws, wout_bf, out);
}
